// InvariancePropagationLoss_86646670229636
// MI455X (gfx1250) — matrix operand order repaired and checked by reading, not compiled, not run on hardware
//
#include <hip/hip_runtime.h>

typedef __attribute__((ext_vector_type(16))) __bf16 v16bf;
typedef __attribute__((ext_vector_type(8)))  __bf16 v8bf;
typedef __attribute__((ext_vector_type(8)))  float  v8f;

#define PCNT 2048          // prototypes
#define NPTS 131072        // memory points
#define DIM  128           // feature dim (K)
#define ZNN  8             // top-k
#define NSPLIT 16          // N-dimension split for parallelism
#define NSEG  (NPTS / NSPLIT)        // 8192 points per segment
#define CAND  (NSPLIT * 16 * 8)      // 2048 candidates per prototype

// ---------------- f32 -> bf16 conversion (prep) ----------------
__global__ void cvt_bf16_kernel(const float* __restrict__ src,
                                __bf16* __restrict__ dst, int n) {
  int i = blockIdx.x * blockDim.x + threadIdx.x;
  if (i < n) dst[i] = (__bf16)src[i];
}

// ---------------- WMMA scoring + per-lane register top-8 ----------------
// One wave per (16-prototype M-tile, N-segment). NT GEMM:
// score[m][n] = sum_k proto[m][k] * mem[n][k]; B columns == mem rows.
__global__ __launch_bounds__(32)
void score_topk_kernel(const __bf16* __restrict__ protoB,
                       const __bf16* __restrict__ memB,
                       float* __restrict__ cand_v,
                       int*   __restrict__ cand_i) {
  const int lane = threadIdx.x;
  const int l16  = lane & 15;
  const int hf   = lane >> 4;     // lane half (0: lanes 0-15, 1: lanes 16-31)
  const int mtile = blockIdx.x;   // 0..PCNT/16-1
  const int seg   = blockIdx.y;   // 0..NSPLIT-1

  // Preload A fragments for all 4 K-chunks of 32.
  // A k order chosen here (matching B): lane half 0 holds K=0..15, half 1 K=16..31
  // per chunk => two contiguous 8-element (16B) loads, concatenated lo|hi.
  v16bf a[4];
  {
    const __bf16* arow = protoB + (size_t)(mtile * 16 + l16) * DIM;
#pragma unroll
    for (int kc = 0; kc < 4; ++kc) {
      v8bf lo = *(const v8bf*)(arow + kc * 32 + 16 * hf);
      v8bf hi = *(const v8bf*)(arow + kc * 32 + 16 * hf + 8);
#pragma unroll
      for (int i = 0; i < 8; ++i) { a[kc][i] = lo[i]; a[kc][i + 8] = hi[i]; }
    }
  }

  // Per-lane top-8 min-replacement lists, one per C row this lane owns.
  float tv[8][8]; int ti[8][8]; float tmin[8]; int tslot[8];
#pragma unroll
  for (int r = 0; r < 8; ++r) {
#pragma unroll
    for (int s = 0; s < 8; ++s) { tv[r][s] = -__builtin_inff(); ti[r][s] = 0; }
    tmin[r] = -__builtin_inff(); tslot[r] = 0;
  }

  const int nbase = seg * NSEG;
  for (int t = 0; t < NSEG / 16; ++t) {
    const int n0 = nbase + t * 16;
    // B 32x16 layout: lane = column (mem row), K = 16*hf .. 16*hf+15 contiguous.
    const __bf16* bp = memB + (size_t)(n0 + l16) * DIM + 16 * hf;
    __builtin_prefetch((const void*)(bp + 16 * DIM), 0, 1);  // next tile
    v16bf b0 = *(const v16bf*)(bp +  0);
    v16bf b1 = *(const v16bf*)(bp + 32);
    v16bf b2 = *(const v16bf*)(bp + 64);
    v16bf b3 = *(const v16bf*)(bp + 96);

    v8f c = {};
    c = __builtin_amdgcn_wmma_f32_16x16x32_bf16(false, a[0], false, b0, (short)0, c, false, false);
    c = __builtin_amdgcn_wmma_f32_16x16x32_bf16(false, a[1], false, b1, (short)0, c, false, false);
    c = __builtin_amdgcn_wmma_f32_16x16x32_bf16(false, a[2], false, b2, (short)0, c, false, false);
    c = __builtin_amdgcn_wmma_f32_16x16x32_bf16(false, a[3], false, b3, (short)0, c, false, false);

    const int n = n0 + l16;  // column this lane owns in C
#pragma unroll
    for (int r = 0; r < 8; ++r) {
      float v = c[r];
      if (v > tmin[r]) {                 // rare after warm-up
        int sl = tslot[r];
#pragma unroll
        for (int s = 0; s < 8; ++s)      // constant-indexed: stays in VGPRs
          if (s == sl) { tv[r][s] = v; ti[r][s] = n; }
        float mn = tv[r][0]; int ms = 0;
#pragma unroll
        for (int s = 1; s < 8; ++s)
          if (tv[r][s] < mn) { mn = tv[r][s]; ms = s; }
        tmin[r] = mn; tslot[r] = ms;
      }
    }
  }

  // Spill candidates: [proto][seg][lane16][8]
#pragma unroll
  for (int r = 0; r < 8; ++r) {
    const int proto = mtile * 16 + (r + 8 * hf);
    const size_t base = ((size_t)proto * NSPLIT + seg) * 128 + (size_t)l16 * 8;
#pragma unroll
    for (int s = 0; s < 8; ++s) {
      cand_v[base + s] = tv[r][s];
      cand_i[base + s] = ti[r][s];
    }
  }
}

// ---------------- merge top-8 + pseudo-prototype MSE partial ----------------
__global__ __launch_bounds__(256)
void merge_kernel(const float* __restrict__ cand_v, const int* __restrict__ cand_i,
                  const float* __restrict__ protoF, const float* __restrict__ memF,
                  float* __restrict__ partials) {
  __shared__ float sv[CAND];
  __shared__ int   si[CAND];
  __shared__ float rv[256];
  __shared__ int   rk[256];
  __shared__ int   chosen[ZNN];
  const int tid = threadIdx.x;
  const int p   = blockIdx.x;

  for (int k = tid; k < CAND; k += 256) {
    sv[k] = cand_v[(size_t)p * CAND + k];
    si[k] = cand_i[(size_t)p * CAND + k];
  }
  __syncthreads();

  for (int z = 0; z < ZNN; ++z) {
    float best = -__builtin_inff(); int bk = 0;
    for (int k = tid; k < CAND; k += 256)
      if (sv[k] > best) { best = sv[k]; bk = k; }
    rv[tid] = best; rk[tid] = bk;
    __syncthreads();
    for (int s = 128; s > 0; s >>= 1) {
      if (tid < s && rv[tid + s] > rv[tid]) { rv[tid] = rv[tid + s]; rk[tid] = rk[tid + s]; }
      __syncthreads();
    }
    if (tid == 0) { chosen[z] = si[rk[0]]; sv[rk[0]] = -__builtin_inff(); }
    __syncthreads();
  }

  float partial = 0.f;
  if (tid < DIM) {
    float s = 0.f;
#pragma unroll
    for (int z = 0; z < ZNN; ++z) s += memF[(size_t)chosen[z] * DIM + tid];
    float e = s * (1.f / ZNN) - protoF[(size_t)p * DIM + tid];
    partial = e * e;
  }
  rv[tid] = partial;
  __syncthreads();
  for (int s = 128; s > 0; s >>= 1) {
    if (tid < s) rv[tid] += rv[tid + s];
    __syncthreads();
  }
  if (tid == 0) partials[p] = rv[0];
}

// ---------------- deterministic final reduction ----------------
__global__ __launch_bounds__(256)
void final_reduce_kernel(const float* __restrict__ partials, float* __restrict__ out) {
  __shared__ float r[256];
  float s = 0.f;
  for (int i = threadIdx.x; i < PCNT; i += 256) s += partials[i];
  r[threadIdx.x] = s;
  __syncthreads();
  for (int st = 128; st > 0; st >>= 1) {
    if (threadIdx.x < st) r[threadIdx.x] += r[threadIdx.x + st];
    __syncthreads();
  }
  if (threadIdx.x == 0) out[0] = r[0] * (1.f / ((float)PCNT * (float)DIM));
}

extern "C" void kernel_launch(void* const* d_in, const int* in_sizes, int n_in,
                              void* d_out, int out_size, void* d_ws, size_t ws_size,
                              hipStream_t stream) {
  const float* protoF = (const float*)d_in[0];   // [2048,128] f32
  const float* memF   = (const float*)d_in[1];   // [131072,128] f32
  float* out = (float*)d_out;

  char* ws = (char*)d_ws;
  __bf16* protoB  = (__bf16*)ws;
  __bf16* memB    = (__bf16*)(ws + (size_t)PCNT * DIM * 2);
  float*  cand_v  = (float*) (ws + (size_t)(PCNT + NPTS) * DIM * 2);
  int*    cand_i  = (int*)   ((char*)cand_v + (size_t)PCNT * CAND * 4);
  float*  partials= (float*) ((char*)cand_i + (size_t)PCNT * CAND * 4);

  cvt_bf16_kernel<<<(PCNT * DIM + 255) / 256, 256, 0, stream>>>(protoF, protoB, PCNT * DIM);
  cvt_bf16_kernel<<<(NPTS * DIM + 255) / 256, 256, 0, stream>>>(memF, memB, NPTS * DIM);
  score_topk_kernel<<<dim3(PCNT / 16, NSPLIT), 32, 0, stream>>>(protoB, memB, cand_v, cand_i);
  merge_kernel<<<PCNT, 256, 0, stream>>>(cand_v, cand_i, protoF, memF, partials);
  final_reduce_kernel<<<1, 256, 0, stream>>>(partials, out);
}
